// PhaseInteractionLayer_73186242724047
// MI455X (gfx1250) — matrix operand order repaired and checked by reading, not compiled, not run on hardware
//
#include <hip/hip_runtime.h>

// ---------------- types ----------------
typedef __bf16 bf16x4  __attribute__((ext_vector_type(4)));
typedef __bf16 bf16x8  __attribute__((ext_vector_type(8)));
typedef __bf16 bf16x16 __attribute__((ext_vector_type(16)));
typedef float  f32x8   __attribute__((ext_vector_type(8)));

#define H_DIM  1024
#define TWO_H  2048
#define B_DIM  8
#define S_DIM  4096
#define M_ROWS (B_DIM * S_DIM)   // 32768
#define EPS_F  1e-5f
#define MIXSC  0.1f

// ---------------- fp32 -> bf16 flat convert (x) ----------------
__global__ __launch_bounds__(256)
void cvt_f32_bf16_kernel(const float* __restrict__ in, __bf16* __restrict__ out)
{
    size_t i = ((size_t)blockIdx.x * 256 + threadIdx.x) * 4;
    float4 f = *(const float4*)(in + i);
    bf16x4 r;
    r[0] = (__bf16)f.x; r[1] = (__bf16)f.y; r[2] = (__bf16)f.z; r[3] = (__bf16)f.w;
    *(bf16x4*)(out + i) = r;
}

// ---------------- fp32 [K,N] -> bf16 [N,K] transpose-convert (weights) ----
__global__ __launch_bounds__(256)
void cvt_transpose_kernel(const float* __restrict__ W, __bf16* __restrict__ Wt,
                          int K, int N)
{
    __shared__ __align__(16) __bf16 lT[32][40];
    const int tid = threadIdx.x;
    const int n0 = blockIdx.x * 32;
    const int k0 = blockIdx.y * 32;
    {
        const int k  = tid >> 3;
        const int n4 = (tid & 7) * 4;
        float4 f = *(const float4*)(W + (size_t)(k0 + k) * N + (n0 + n4));
        lT[n4 + 0][k] = (__bf16)f.x;
        lT[n4 + 1][k] = (__bf16)f.y;
        lT[n4 + 2][k] = (__bf16)f.z;
        lT[n4 + 3][k] = (__bf16)f.w;
    }
    __syncthreads();
    {
        const int n  = tid >> 3;
        const int kq = (tid & 7) * 4;
        bf16x4 v = *(const bf16x4*)&lT[n][kq];
        *(bf16x4*)(Wt + (size_t)(n0 + n) * K + (k0 + kq)) = v;
    }
}

// ---------------- WMMA GEMM (all-bf16 operands) ----------------
// C[M,N] = A[M,K] * Bt[N,K]^T + bias[N]
// Block = 256 threads = 8 wave32. Block tile 64(M) x 128(N). K step 64.
// Waves: msub = w>>2 (0..1) -> rows msub*32..+31 ; nsub = w&3 -> cols nsub*32..+31.
// Wave tile 32x32: 2 A-frags x 2 B-frags -> 4 wmma per K-subchunk (8 per K-step),
// every ds_load_b128 feeds two v_wmma.
__global__ __launch_bounds__(256)
void wmma_gemm_kernel(const __bf16* __restrict__ A, const __bf16* __restrict__ Bt,
                      const float* __restrict__ bias, float* __restrict__ C,
                      int M, int N, int K)
{
    __shared__ __align__(16) __bf16 lA[64][72];   // [m][k], 144B rows
    __shared__ __align__(16) __bf16 lB[128][72];  // [n][k]

    const int tid  = threadIdx.x;
    const int lane = tid & 31;
    const int wave = tid >> 5;
    const int msub = wave >> 2;         // 0..1
    const int nsub = wave & 3;          // 0..3
    const int half = lane >> 4;         // 0/1
    const int r    = lane & 15;

    const int n0 = blockIdx.x * 128;
    const int m0 = blockIdx.y * 64;

    // staging: A 64x64 -> 16 bf16/thread (2x b128); B 128x64 -> 32 bf16/thread (4x b128)
    const int sa_row = tid >> 2;        // 0..63
    const int sa_k   = (tid & 3) * 16;  // 0,16,32,48
    const int sb_row = tid >> 1;        // 0..127
    const int sb_k   = (tid & 1) * 32;  // 0,32

    f32x8 acc00 = {}, acc01 = {}, acc10 = {}, acc11 = {};

    const int a_row0 = msub * 32 + r;
    const int a_row1 = a_row0 + 16;
    const int b_row0 = nsub * 32 + r;
    const int b_row1 = b_row0 + 16;

    for (int k0 = 0; k0 < K; k0 += 64) {
        // ---- stage tiles ----
        {
            const __bf16* ap = A  + (size_t)(m0 + sa_row) * K + (k0 + sa_k);
            *(bf16x8*)&lA[sa_row][sa_k]     = *(const bf16x8*)(ap);
            *(bf16x8*)&lA[sa_row][sa_k + 8] = *(const bf16x8*)(ap + 8);
            const __bf16* bp = Bt + (size_t)(n0 + sb_row) * K + (k0 + sb_k);
            *(bf16x8*)&lB[sb_row][sb_k]      = *(const bf16x8*)(bp);
            *(bf16x8*)&lB[sb_row][sb_k + 8]  = *(const bf16x8*)(bp + 8);
            *(bf16x8*)&lB[sb_row][sb_k + 16] = *(const bf16x8*)(bp + 16);
            *(bf16x8*)&lB[sb_row][sb_k + 24] = *(const bf16x8*)(bp + 24);
        }
        if (k0 + 64 < K) {  // global_prefetch_b8 next chunk
            __builtin_prefetch(A  + (size_t)(m0 + sa_row) * K + (k0 + 64 + sa_k), 0, 1);
            __builtin_prefetch(Bt + (size_t)(n0 + sb_row) * K + (k0 + 64 + sb_k), 0, 1);
        }
        __syncthreads();

        // ---- 2 K-subchunks of 32; fragments per ISA 7.12.2 ----
        #pragma unroll
        for (int kk = 0; kk < 64; kk += 32) {
            union { bf16x16 v16; bf16x8 v8[2]; } a0, a1, b0, b1;
            // A 16x32: lane(r,half): row r; e0..7 -> K=half*8+e, e8..15 -> 16+half*8+e
            a0.v8[0] = *(const bf16x8*)&lA[a_row0][kk + half * 8];
            a0.v8[1] = *(const bf16x8*)&lA[a_row0][kk + 16 + half * 8];
            a1.v8[0] = *(const bf16x8*)&lA[a_row1][kk + half * 8];
            a1.v8[1] = *(const bf16x8*)&lA[a_row1][kk + 16 + half * 8];
            // B 32x16: lane(r,half): col r; e0..7 -> K=half*8+e, e8..15 -> 16+half*8+e
            b0.v8[0] = *(const bf16x8*)&lB[b_row0][kk + half * 8];
            b0.v8[1] = *(const bf16x8*)&lB[b_row0][kk + 16 + half * 8];
            b1.v8[0] = *(const bf16x8*)&lB[b_row1][kk + half * 8];
            b1.v8[1] = *(const bf16x8*)&lB[b_row1][kk + 16 + half * 8];

            acc00 = __builtin_amdgcn_wmma_f32_16x16x32_bf16(
                        false, a0.v16, false, b0.v16, (short)0, acc00, false, false);
            acc01 = __builtin_amdgcn_wmma_f32_16x16x32_bf16(
                        false, a0.v16, false, b1.v16, (short)0, acc01, false, false);
            acc10 = __builtin_amdgcn_wmma_f32_16x16x32_bf16(
                        false, a1.v16, false, b0.v16, (short)0, acc10, false, false);
            acc11 = __builtin_amdgcn_wmma_f32_16x16x32_bf16(
                        false, a1.v16, false, b1.v16, (short)0, acc11, false, false);
        }
        __syncthreads();
    }

    // ---- epilogue: C/D layout — lane(r,half): col=r, rows half*8+v ----
    const int col0 = n0 + nsub * 32 + r;
    const int col1 = col0 + 16;
    const float bb0 = bias[col0];
    const float bb1 = bias[col1];
    const int rb0 = m0 + msub * 32 + half * 8;
    const int rb1 = rb0 + 16;
    #pragma unroll
    for (int v = 0; v < 8; ++v) {
        C[(size_t)(rb0 + v) * N + col0] = acc00[v] + bb0;
        C[(size_t)(rb0 + v) * N + col1] = acc01[v] + bb1;
        C[(size_t)(rb1 + v) * N + col0] = acc10[v] + bb0;
        C[(size_t)(rb1 + v) * N + col1] = acc11[v] + bb1;
    }
}

// ---------------- per-(b,h) circular stats: sum cos(phase), sum sin(phase) ----
__global__ __launch_bounds__(256)
void phase_stats_kernel(const float* __restrict__ cx,
                        float* __restrict__ sumc, float* __restrict__ sums)
{
    const int h  = blockIdx.x * 256 + threadIdx.x;
    const int s0 = blockIdx.y * 128;
    const int b  = blockIdx.z;
    float ac = 0.f, as = 0.f;
    for (int s = 0; s < 128; ++s) {
        size_t base = ((size_t)b * S_DIM + (s0 + s)) * TWO_H;
        float re = cx[base + h];
        float im = cx[base + H_DIM + h];
        float rm = rsqrtf(re * re + im * im + 1e-30f);
        ac += re * rm;
        as += im * rm;
    }
    atomicAdd(&sumc[b * H_DIM + h], ac);
    atomicAdd(&sums[b * H_DIM + h], as);
}

// ---------------- mixing + rotation -> bf16 new_complex ----------------
__global__ __launch_bounds__(256)
void mix_kernel(const float* __restrict__ cx, const float* __restrict__ sumc,
                const float* __restrict__ sums, const float* __restrict__ shifts,
                __bf16* __restrict__ outc)
{
    const int h = blockIdx.x * 256 + threadIdx.x;
    const int s = blockIdx.y;
    const int b = blockIdx.z;
    size_t base = ((size_t)b * S_DIM + s) * TWO_H;
    float re = cx[base + h];
    float im = cx[base + H_DIM + h];
    float mag = sqrtf(re * re + im * im + EPS_F);
    float rm  = rsqrtf(re * re + im * im + 1e-30f);
    float cph = re * rm, sph = im * rm;
    float sh = shifts[h];
    float ssh, csh; __sincosf(sh, &ssh, &csh);
    float mc = sumc[b * H_DIM + h] * (1.0f / S_DIM);
    float ms = sums[b * H_DIM + h] * (1.0f / S_DIM);
    float mixed = (cph * csh - sph * ssh) * mc + (sph * csh + cph * ssh) * ms;
    float d = mixed * MIXSC;
    float sd, cd; __sincosf(d, &sd, &cd);
    float nr = mag * (cph * cd - sph * sd);
    float ni = mag * (sph * cd + cph * sd);
    outc[base + h]         = (__bf16)nr;
    outc[base + H_DIM + h] = (__bf16)ni;
}

// ---------------- launch ----------------
extern "C" void kernel_launch(void* const* d_in, const int* in_sizes, int n_in,
                              void* d_out, int out_size, void* d_ws, size_t ws_size,
                              hipStream_t stream)
{
    const float* x    = (const float*)d_in[0];
    const float* Wto  = (const float*)d_in[1];
    const float* bto  = (const float*)d_in[2];
    const float* Wfr  = (const float*)d_in[3];
    const float* bfr  = (const float*)d_in[4];
    const float* shf  = (const float*)d_in[5];
    float*       out  = (float*)d_out;

    char* ws = (char*)d_ws;
    size_t off = 0;
    float*  cx    = (float*)(ws + off);  off += (size_t)M_ROWS * TWO_H * sizeof(float);   // 268 MB
    __bf16* newc  = (__bf16*)(ws + off); off += (size_t)M_ROWS * TWO_H * sizeof(__bf16);  // 134 MB
    __bf16* xb    = (__bf16*)(ws + off); off += (size_t)M_ROWS * H_DIM * sizeof(__bf16);  //  67 MB
    __bf16* WtoT  = (__bf16*)(ws + off); off += (size_t)TWO_H * H_DIM * sizeof(__bf16);   //   4 MB
    __bf16* WfrT  = (__bf16*)(ws + off); off += (size_t)H_DIM * TWO_H * sizeof(__bf16);   //   4 MB
    float*  sumc  = (float*)(ws + off);  off += (size_t)B_DIM * H_DIM * sizeof(float);
    float*  sums  = (float*)(ws + off);

    dim3 blk(256);

    // one-time conversions per call
    cvt_f32_bf16_kernel<<<(size_t)M_ROWS * H_DIM / 1024, blk, 0, stream>>>(x, xb);
    cvt_transpose_kernel<<<dim3(TWO_H / 32, H_DIM / 32), blk, 0, stream>>>(Wto, WtoT, H_DIM, TWO_H);
    cvt_transpose_kernel<<<dim3(H_DIM / 32, TWO_H / 32), blk, 0, stream>>>(Wfr, WfrT, TWO_H, H_DIM);

    hipMemsetAsync(sumc, 0, (size_t)2 * B_DIM * H_DIM * sizeof(float), stream);

    // GEMM1: complex = x @ W_to + b_to   (M=32768, N=2048, K=1024)
    wmma_gemm_kernel<<<dim3(TWO_H / 128, M_ROWS / 64), blk, 0, stream>>>(
        xb, WtoT, bto, cx, M_ROWS, TWO_H, H_DIM);
    // circular means over S
    phase_stats_kernel<<<dim3(H_DIM / 256, S_DIM / 128, B_DIM), blk, 0, stream>>>(
        cx, sumc, sums);
    // phase mixing -> bf16 new_complex
    mix_kernel<<<dim3(H_DIM / 256, S_DIM, B_DIM), blk, 0, stream>>>(
        cx, sumc, sums, shf, newc);
    // GEMM2: out = new_complex @ W_from + b_from   (M=32768, N=1024, K=2048)
    wmma_gemm_kernel<<<dim3(H_DIM / 128, M_ROWS / 64), blk, 0, stream>>>(
        newc, WfrT, bfr, out, M_ROWS, H_DIM, TWO_H);
}
